// MultiHeadAttention_66408784330989
// MI455X (gfx1250) — compile-verified, matrix operand order checked by reading, not run on hardware
//
#include <hip/hip_runtime.h>
#include <hip/hip_bf16.h>
#include <stdint.h>

// ---------------------------------------------------------------------------
// MI455X (gfx1250) multi-head attention.
//  - all matmuls via v_wmma_f32_16x16x32_bf16 (f32 accumulate)
//  - flash-style fused softmax (S x S scores never reach HBM)
//  - weight tiles staged to LDS by the Tensor Data Mover (tensor_load_to_lds),
//    double buffered, shared by all 8 waves of a workgroup
//  - unconditional software pipelining (no PHI copies), k-loops unrolled x2
// ---------------------------------------------------------------------------

typedef __bf16 bf16_t;
typedef __attribute__((ext_vector_type(16))) __bf16 v16bf;
typedef __attribute__((ext_vector_type(8)))  __bf16 v8bf;
typedef __attribute__((ext_vector_type(8)))  float  v8f;
typedef __attribute__((ext_vector_type(4)))  unsigned int u32x4;
typedef __attribute__((ext_vector_type(8)))  unsigned int u32x8;

#define SEQ   4096
#define EMBED 1024
#define HEADS 16
#define DH    64

// ----- WMMA helpers ---------------------------------------------------------
// 16-bit A-matrix layout (ISA 7.12.2): lane L holds row M = L%16; elements
// cover K = [8*(L/16), +8) and K = [16+8*(L/16), +8) -> two 16B loads/lane.
// B fragments load columns of B (rows of B^T) with the identical pattern.
__device__ __forceinline__ v16bf load_frag(const bf16_t* __restrict__ base, int ld) {
  const int lane = threadIdx.x & 31;
  const int r    = lane & 15;
  const int kb   = (lane >> 4) << 3;
  const bf16_t* p = base + (size_t)r * ld + kb;
  v8bf lo = *reinterpret_cast<const v8bf*>(p);
  v8bf hi = *reinterpret_cast<const v8bf*>(p + 16);
  v16bf out;
#pragma unroll
  for (int i = 0; i < 8; ++i) { out[i] = lo[i]; out[i + 8] = hi[i]; }
  return out;
}

__device__ __forceinline__ v8f wmma_bf16(v16bf a, v16bf b, v8f c) {
  return __builtin_amdgcn_wmma_f32_16x16x32_bf16(false, a, false, b, (short)0, c,
                                                 false, false);
}
// C/D layout: c[i] = element (m = 8*(lane>>4) + i, n = lane&15)

// ----- Tensor Data Mover: load a 64-row x 32-col bf16 tile into LDS ---------
// D# per CDNA5 ISA. 2D tile, data_size=2B,
// row stride = EMBED elements. Inline asm form is toolchain-portable.
__device__ __forceinline__ unsigned lds_addr_of(const void* p) {
  return (unsigned)(uintptr_t)p;   // generic ptr low 32 bits == LDS byte address
}

__device__ __forceinline__ void tdm_load_w_tile(unsigned ldsaddr, const bf16_t* gsrc) {
  unsigned long long ga = (unsigned long long)(uintptr_t)gsrc;
  u32x4 g0;
  u32x8 g1;
  g0[0] = 1u;                                                   // count=1
  g0[1] = ldsaddr;                                              // lds_addr
  g0[2] = (unsigned)ga;                                         // global_addr lo
  g0[3] = (unsigned)((ga >> 32) & 0x1FFFFFFull) | (2u << 30);   // addr[56:32]|type=2
  g1[0] = 1u << 16;                                             // data_size=1 (2B)
  g1[1] = (unsigned)((EMBED & 0xFFFFu) << 16);                  // tensor_dim0 lo16
  g1[2] = (unsigned)((EMBED >> 16) | (64u << 16));              // dim0 hi | tensor_dim1=64
  g1[3] = (32u << 16);                                          // dim1 hi=0 | tile_dim0=32
  g1[4] = 64u;                                                  // tile_dim1=64, tile_dim2=0
  g1[5] = (unsigned)EMBED;                                      // tensor_dim0_stride lo
  g1[6] = 0u;
  g1[7] = 0u;
  asm volatile("tensor_load_to_lds %0, %1" :: "s"(g0), "s"(g1) : "memory");
}
__device__ __forceinline__ void tdm_wait() {
  __builtin_amdgcn_s_wait_tensorcnt((short)0);   // free for waves with cnt==0
}

// ----- data-prep kernels ----------------------------------------------------
__global__ void k_cvt_x(const float* __restrict__ x, bf16_t* __restrict__ xb, int n) {
  int i = blockIdx.x * blockDim.x + threadIdx.x;
  if (i < n) xb[i] = (bf16_t)x[i];
}

// src f32 [B, R, C] row-major -> dst bf16 [B, C, R]
__global__ void k_transpose_w(const float* __restrict__ src, bf16_t* __restrict__ dst,
                              int R, int C) {
  int b = blockIdx.y;
  int i = blockIdx.x * blockDim.x + threadIdx.x;
  if (i >= R * C) return;
  int r = i / C, c = i - r * C;
  dst[(size_t)b * R * C + (size_t)c * R + r] = (bf16_t)src[(size_t)b * R * C + i];
}

// ----- QKV projection -------------------------------------------------------
// wt: [H, DH, EMBED] pre-transposed. Each wave: 32x64 tile; 8 waves = 256 rows.
// Weight tile (64x32) comes from LDS, TDM double-buffered.
__global__ __launch_bounds__(256)
void k_qkv(const bf16_t* __restrict__ xb, const bf16_t* __restrict__ wt,
           const float* __restrict__ bias, bf16_t* __restrict__ outRM,
           bf16_t* __restrict__ outT, int transposed) {
  __shared__ bf16_t bsm[2][64 * 32];
  const int h    = blockIdx.y;
  const int wid  = threadIdx.x >> 5;
  const int lane = threadIdx.x & 31;
  const int ln = lane & 15, hh = lane >> 4;
  const int m0 = blockIdx.x * 256 + wid * 32;
  const bf16_t* wth = wt + (size_t)h * DH * EMBED;

  if (wid == 0) tdm_load_w_tile(lds_addr_of(&bsm[0][0]), wth);
  tdm_wait();
  __syncthreads();

  v8f c0[4] = {}, c1[4] = {};
  v16bf a0 = load_frag(xb + (size_t)m0 * EMBED, EMBED);
  v16bf a1 = load_frag(xb + (size_t)(m0 + 16) * EMBED, EMBED);

#pragma unroll 2
  for (int kt = 0; kt < EMBED / 32; ++kt) {
    // prefetch next tiles unconditionally (one-iteration overshoot stays in ws)
    if (wid == 0)
      tdm_load_w_tile(lds_addr_of(&bsm[(kt + 1) & 1][0]), wth + (kt + 1) * 32);
    v16bf a0n = load_frag(xb + (size_t)m0 * EMBED + (kt + 1) * 32, EMBED);
    v16bf a1n = load_frag(xb + (size_t)(m0 + 16) * EMBED + (kt + 1) * 32, EMBED);

    const bf16_t* bb = &bsm[kt & 1][0];
#pragma unroll
    for (int j = 0; j < 4; ++j) {
      v16bf b = load_frag(bb + j * 16 * 32, 32);
      c0[j] = wmma_bf16(a0, b, c0[j]);
      c1[j] = wmma_bf16(a1, b, c1[j]);
    }
    tdm_wait();
    __syncthreads();
    a0 = a0n; a1 = a1n;
  }

#pragma unroll
  for (int t = 0; t < 2; ++t) {
    v8f* c = t ? c1 : c0;
    const int mb = m0 + t * 16;
#pragma unroll
    for (int j = 0; j < 4; ++j) {
      float bv = bias[h * DH + j * 16 + ln];
      if (transposed) {
        v8bf pk;
#pragma unroll
        for (int i = 0; i < 8; ++i) pk[i] = (bf16_t)(c[j][i] + bv);
        *reinterpret_cast<v8bf*>(outT + ((size_t)h * DH + j * 16 + ln) * SEQ + mb + 8 * hh) = pk;
      } else {
#pragma unroll
        for (int i = 0; i < 8; ++i)
          outRM[((size_t)h * SEQ + mb + 8 * hh + i) * DH + j * 16 + ln] =
              (bf16_t)(c[j][i] + bv);
      }
    }
  }
}

// ----- fused flash attention ------------------------------------------------
__global__ __launch_bounds__(256)
void k_attn(const bf16_t* __restrict__ Qb, const bf16_t* __restrict__ Kb,
            const bf16_t* __restrict__ Vt, bf16_t* __restrict__ xa) {
  __shared__ bf16_t plds[8][16 * 32];
  const int h    = blockIdx.y;
  const int wid  = threadIdx.x >> 5;
  const int lane = threadIdx.x & 31;
  const int ln = lane & 15, hh = lane >> 4;
  const int q0 = blockIdx.x * 128 + wid * 16;

  const bf16_t* Qh = Qb + (size_t)h * SEQ * DH;
  const bf16_t* Kh = Kb + (size_t)h * SEQ * DH;
  const bf16_t* Vh = Vt + (size_t)h * DH * SEQ;

  v16bf aq0 = load_frag(Qh + (size_t)q0 * DH + 0,  DH);
  v16bf aq1 = load_frag(Qh + (size_t)q0 * DH + 32, DH);

  float mrow[8], lrow[8];
#pragma unroll
  for (int i = 0; i < 8; ++i) { mrow[i] = -3.0e38f; lrow[i] = 0.0f; }
  v8f o[4] = {};
  bf16_t* pw = &plds[wid][0];

  v16bf bk[4];
  bk[0] = load_frag(Kh + 0,            DH);
  bk[1] = load_frag(Kh + 32,           DH);
  bk[2] = load_frag(Kh + 16 * DH + 0,  DH);
  bk[3] = load_frag(Kh + 16 * DH + 32, DH);

#pragma unroll 2
  for (int t0 = 0; t0 < SEQ; t0 += 32) {
    v8f s0 = {}, s1 = {};
    s0 = wmma_bf16(aq0, bk[0], s0); s0 = wmma_bf16(aq1, bk[1], s0);
    s1 = wmma_bf16(aq0, bk[2], s1); s1 = wmma_bf16(aq1, bk[3], s1);

    // prefetch V (this block) + next K; latency hides under softmax VALU
    v16bf bv[4];
#pragma unroll
    for (int j = 0; j < 4; ++j)
      bv[j] = load_frag(Vh + (size_t)(j * 16) * SEQ + t0, SEQ);
    {
      const bf16_t* Kn = Kh + (size_t)(t0 + 32) * DH;   // overshoot stays in ws
      bk[0] = load_frag(Kn + 0,            DH);
      bk[1] = load_frag(Kn + 32,           DH);
      bk[2] = load_frag(Kn + 16 * DH + 0,  DH);
      bk[3] = load_frag(Kn + 16 * DH + 32, DH);
    }

#pragma unroll
    for (int i = 0; i < 8; ++i) {
      float x0 = s0[i] * 0.125f, x1 = s1[i] * 0.125f;   // 1/sqrt(64)
      float mx = fmaxf(x0, x1);
      mx = fmaxf(mx, __shfl_xor(mx, 1, 32));
      mx = fmaxf(mx, __shfl_xor(mx, 2, 32));
      mx = fmaxf(mx, __shfl_xor(mx, 4, 32));
      mx = fmaxf(mx, __shfl_xor(mx, 8, 32));            // row lives in a 16-lane half
      float mnew = fmaxf(mrow[i], mx);
      float p0 = __expf(x0 - mnew), p1 = __expf(x1 - mnew);
      float rs = p0 + p1;
      rs += __shfl_xor(rs, 1, 32);
      rs += __shfl_xor(rs, 2, 32);
      rs += __shfl_xor(rs, 4, 32);
      rs += __shfl_xor(rs, 8, 32);
      float alpha = __expf(mrow[i] - mnew);
      lrow[i] = lrow[i] * alpha + rs;
      mrow[i] = mnew;
#pragma unroll
      for (int j = 0; j < 4; ++j) o[j][i] *= alpha;
      pw[(8 * hh + i) * 32 + ln]      = (bf16_t)p0;
      pw[(8 * hh + i) * 32 + 16 + ln] = (bf16_t)p1;
    }
    v16bf ap = load_frag(pw, 32);                       // C-layout -> A-layout via LDS
#pragma unroll
    for (int j = 0; j < 4; ++j) o[j] = wmma_bf16(ap, bv[j], o[j]);
  }
#pragma unroll
  for (int i = 0; i < 8; ++i) {
    float inv = 1.0f / lrow[i];
#pragma unroll
    for (int j = 0; j < 4; ++j)
      xa[(size_t)(q0 + 8 * hh + i) * EMBED + h * DH + j * 16 + ln] =
          (bf16_t)(o[j][i] * inv);
  }
}

// ----- output projection ----------------------------------------------------
__global__ __launch_bounds__(256)
void k_oproj(const bf16_t* __restrict__ xa, const bf16_t* __restrict__ wot,
             const float* __restrict__ bo, float* __restrict__ out) {
  __shared__ bf16_t bsm[2][64 * 32];
  const int wid  = threadIdx.x >> 5;
  const int lane = threadIdx.x & 31;
  const int ln = lane & 15, hh = lane >> 4;
  const int m0 = blockIdx.x * 256 + wid * 32;
  const int n0 = blockIdx.y * 64;
  const bf16_t* wbase = wot + (size_t)n0 * EMBED;

  if (wid == 0) tdm_load_w_tile(lds_addr_of(&bsm[0][0]), wbase);
  tdm_wait();
  __syncthreads();

  v8f c0[4] = {}, c1[4] = {};
  v16bf a0 = load_frag(xa + (size_t)m0 * EMBED, EMBED);
  v16bf a1 = load_frag(xa + (size_t)(m0 + 16) * EMBED, EMBED);

#pragma unroll 2
  for (int kt = 0; kt < EMBED / 32; ++kt) {
    if (wid == 0)
      tdm_load_w_tile(lds_addr_of(&bsm[(kt + 1) & 1][0]), wbase + (kt + 1) * 32);
    v16bf a0n = load_frag(xa + (size_t)m0 * EMBED + (kt + 1) * 32, EMBED);
    v16bf a1n = load_frag(xa + (size_t)(m0 + 16) * EMBED + (kt + 1) * 32, EMBED);

    const bf16_t* bb = &bsm[kt & 1][0];
#pragma unroll
    for (int j = 0; j < 4; ++j) {
      v16bf b = load_frag(bb + j * 16 * 32, 32);
      c0[j] = wmma_bf16(a0, b, c0[j]);
      c1[j] = wmma_bf16(a1, b, c1[j]);
    }
    tdm_wait();
    __syncthreads();
    a0 = a0n; a1 = a1n;
  }

#pragma unroll
  for (int t = 0; t < 2; ++t) {
    v8f* c = t ? c1 : c0;
    const int mb = m0 + t * 16;
#pragma unroll
    for (int j = 0; j < 4; ++j) {
      float bv = bo[n0 + j * 16 + ln];
#pragma unroll
      for (int i = 0; i < 8; ++i)
        out[((size_t)mb + 8 * hh + i) * EMBED + n0 + j * 16 + ln] = c[j][i] + bv;
    }
  }
}

// ---------------------------------------------------------------------------
extern "C" void kernel_launch(void* const* d_in, const int* in_sizes, int n_in,
                              void* d_out, int out_size, void* d_ws, size_t ws_size,
                              hipStream_t stream) {
  (void)in_sizes; (void)n_in; (void)out_size; (void)ws_size;
  const float* x  = (const float*)d_in[0];
  const float* Wq = (const float*)d_in[1];
  const float* bq = (const float*)d_in[2];
  const float* Wk = (const float*)d_in[3];
  const float* bk = (const float*)d_in[4];
  const float* Wv = (const float*)d_in[5];
  const float* bv = (const float*)d_in[6];
  const float* Wo = (const float*)d_in[7];
  const float* bo = (const float*)d_in[8];
  float* out = (float*)d_out;

  char*  ws  = (char*)d_ws;
  size_t off = 0;
  auto alloc = [&](size_t bytes) {
    void* p = ws + off;
    off += (bytes + 255) & ~(size_t)255;
    return p;
  };
  bf16_t* xb  = (bf16_t*)alloc((size_t)SEQ * EMBED * 2);
  bf16_t* Qb  = (bf16_t*)alloc((size_t)HEADS * SEQ * DH * 2);
  bf16_t* Kb  = (bf16_t*)alloc((size_t)HEADS * SEQ * DH * 2);
  bf16_t* VtB = (bf16_t*)alloc((size_t)HEADS * DH * SEQ * 2);
  bf16_t* xa  = (bf16_t*)alloc((size_t)SEQ * EMBED * 2);
  bf16_t* wtq = (bf16_t*)alloc((size_t)HEADS * DH * EMBED * 2);
  bf16_t* wtk = (bf16_t*)alloc((size_t)HEADS * DH * EMBED * 2);
  bf16_t* wtv = (bf16_t*)alloc((size_t)HEADS * DH * EMBED * 2);
  bf16_t* wot = (bf16_t*)alloc((size_t)EMBED * EMBED * 2);
  (void)alloc(4096);   // pad: absorbs unconditional-prefetch overshoot

  {
    int n = SEQ * EMBED;
    k_cvt_x<<<dim3((n + 255) / 256), dim3(256), 0, stream>>>(x, xb, n);
    k_transpose_w<<<dim3((EMBED * DH + 255) / 256, HEADS), dim3(256), 0, stream>>>(Wq, wtq, EMBED, DH);
    k_transpose_w<<<dim3((EMBED * DH + 255) / 256, HEADS), dim3(256), 0, stream>>>(Wk, wtk, EMBED, DH);
    k_transpose_w<<<dim3((EMBED * DH + 255) / 256, HEADS), dim3(256), 0, stream>>>(Wv, wtv, EMBED, DH);
    k_transpose_w<<<dim3((EMBED * EMBED + 255) / 256, 1), dim3(256), 0, stream>>>(Wo, wot, EMBED, EMBED);
  }

  dim3 gqkv(SEQ / 256, HEADS);
  k_qkv<<<gqkv, dim3(256), 0, stream>>>(xb, wtq, bq, Qb, VtB, 0);
  k_qkv<<<gqkv, dim3(256), 0, stream>>>(xb, wtk, bk, Kb, VtB, 0);
  k_qkv<<<gqkv, dim3(256), 0, stream>>>(xb, wtv, bv, Qb /*unused*/, VtB, 1);

  k_attn<<<dim3(SEQ / 128, HEADS), dim3(256), 0, stream>>>(Qb, Kb, VtB, xa);

  k_oproj<<<dim3(SEQ / 256, EMBED / 64), dim3(256), 0, stream>>>(xa, wot, bo, out);
}
